// LCGraphNet_18210661335123
// MI455X (gfx1250) — matrix operand order repaired and checked by reading, not compiled, not run on hardware
//
#include <hip/hip_runtime.h>
#include <hip/hip_bf16.h>

// ---------------------------------------------------------------------------
// DGCNN-style net on MI455X (gfx1250, wave32).
// - Edge-MLP / head GEMMs: v_wmma_f32_16x16x32_bf16, f32 accumulate.
// - Weights async-staged to LDS (global_load_async_to_lds_b128 + ASYNCcnt),
//   B fragments served by ds_load_b128; A streamed from L2.
// - kNN: f32 VALU distances fused with unordered top-8 (sum-aggregation makes
//   neighbor order irrelevant).
// ---------------------------------------------------------------------------

typedef __attribute__((ext_vector_type(16))) __bf16 v16bf;
typedef __attribute__((ext_vector_type(8)))  float  v8f;

#define NPTS   16384
#define KNN    8
#define NEDGE  (NPTS * KNN)   // 131072

__device__ __forceinline__ unsigned short f2bf(float f) {
    union { float f; unsigned int u; } x; x.f = f;
    unsigned int u = x.u;
    unsigned int r = u + 0x7FFFu + ((u >> 16) & 1u);   // round-to-nearest-even
    return (unsigned short)(r >> 16);
}
__device__ __forceinline__ float bf2f(unsigned short h) {
    union { unsigned int u; float f; } x; x.u = ((unsigned int)h) << 16; return x.f;
}

// ---------------- row squared-norms ----------------------------------------
__global__ void k_norms(const float* __restrict__ x, float* __restrict__ x2,
                        int N, int D) {
    int i = blockIdx.x * blockDim.x + threadIdx.x;
    if (i >= N) return;
    float s = 0.f;
    for (int d = 0; d < D; ++d) { float v = x[(size_t)i * D + d]; s = fmaf(v, v, s); }
    x2[i] = s;
}

// ---------------- kNN: one thread per query row, LDS-tiled candidates ------
template<int D>
__global__ void k_knn(const float* __restrict__ x, const float* __restrict__ x2,
                      int* __restrict__ idx, int N) {
    __shared__ float tile[128 * D];
    __shared__ float tz[128];
    int i = blockIdx.x * blockDim.x + threadIdx.x;   // blockDim = 64
    float xi[D];
#pragma unroll
    for (int d = 0; d < D; ++d) xi[d] = x[(size_t)i * D + d];
    float zi = x2[i];

    float bd[KNN]; int bi[KNN];
#pragma unroll
    for (int k = 0; k < KNN; ++k) { bd[k] = 3.4e38f; bi[k] = 0; }
    float worst = 3.4e38f; int wslot = 0;

    for (int jt = 0; jt < N; jt += 128) {
        for (int t = threadIdx.x; t < 128 * D; t += blockDim.x)
            tile[t] = x[(size_t)jt * D + t];
        for (int t = threadIdx.x; t < 128; t += blockDim.x)
            tz[t] = x2[jt + t];
        __syncthreads();
        for (int jj = 0; jj < 128; ++jj) {
            const float* xj = &tile[jj * D];
            float dot = 0.f;
#pragma unroll
            for (int d = 0; d < D; ++d) dot = fmaf(xi[d], xj[d], dot);
            float dist = zi + tz[jj] - 2.f * dot;
            if (dist < worst) {                 // rare after warm-up
#pragma unroll
                for (int k = 0; k < KNN; ++k)
                    if (k == wslot) { bd[k] = dist; bi[k] = jt + jj; }
                worst = bd[0]; wslot = 0;
#pragma unroll
                for (int k = 1; k < KNN; ++k)
                    if (bd[k] > worst) { worst = bd[k]; wslot = k; }
            }
        }
        __syncthreads();
    }
#pragma unroll
    for (int k = 0; k < KNN; ++k) idx[(size_t)i * KNN + k] = bi[k];
}

// ---------------- edge features [x_i, x_j - x_i] -> bf16 (K-padded) --------
__global__ void k_edge(const float* __restrict__ src, const int* __restrict__ idx,
                       unsigned short* __restrict__ e, int D, int LD) {
    int t = blockIdx.x * blockDim.x + threadIdx.x;
    if (t >= NEDGE * LD) return;
    int r = t / LD, c = t - r * LD;
    int n = r >> 3;                    // K = 8
    float v = 0.f;
    if (c < D) {
        v = src[(size_t)n * D + c];
    } else if (c < 2 * D) {
        int j = idx[r]; int cc = c - D;
        v = src[(size_t)j * D + cc] - src[(size_t)n * D + cc];
    }
    e[t] = f2bf(v);
}

// ---------------- weight pre-swizzle into WMMA B-fragment layout -----------
// wt[((kb*ntn + nb)*32 + lane)*16 + e] = W[kb*32 + 16*(lane>>4) + e][nb*16 + (lane&15)]
__global__ void k_prep_w(const float* __restrict__ w, unsigned short* __restrict__ wt,
                         int Kin, int Kp, int Nout) {
    int t = blockIdx.x * blockDim.x + threadIdx.x;
    if (t >= Kp * Nout) return;
    int e    = t & 15;
    int lane = (t >> 4) & 31;
    int rest = t >> 9;
    int ntn  = Nout >> 4;
    int nb   = rest % ntn;
    int kb   = rest / ntn;
    int n = (nb << 4) + (lane & 15);
    int k = (kb << 5) + ((lane >> 4) << 4) + e;
    float v = (k < Kin) ? w[(size_t)k * Nout + n] : 0.f;
    wt[t] = f2bf(v);
}

// ---------------- bf16 WMMA GEMM -------------------------------------------
// Block = 8 waves = 128-row stripe. Wave w owns m-tile (blockIdx*8+w) and
// accumulates ALL NTN n-tiles in registers: one A-fragment load feeds NTN
// back-to-back WMMAs. Whole (pre-swizzled) weight matrix is async-copied to
// LDS once per block; B fragments come from ds_load_b128 thereafter.
union Frag { v16bf v; uint4 q[2]; };

template<int NTN>
__global__ void k_gemm(const unsigned short* __restrict__ A, int lda,
                       const unsigned short* __restrict__ Wt,
                       float* __restrict__ C, const float* __restrict__ bias,
                       int relu, int Kp) {
    extern __shared__ unsigned short sW[];           // Kp * NTN*16 bf16
    constexpr int Nout = NTN * 16;
    int tid = threadIdx.x;

    // ---- async-stage weight tiles into LDS (ASYNCcnt-tracked) ----
    int nChunks = (Kp * Nout) >> 3;                  // 16B chunks
    for (int t = tid; t < nChunks; t += 256) {
        unsigned    loff = (unsigned)(size_t)(sW + (t << 3));
        const void* g    = (const void*)(Wt + (t << 3));
        asm volatile("global_load_async_to_lds_b128 %0, %1, off"
                     :: "v"(loff), "v"(g) : "memory");
    }
    asm volatile("s_wait_asynccnt 0x0" ::: "memory");
    __syncthreads();

    int wave = tid >> 5, lane = tid & 31;
    int mt  = blockIdx.x * 8 + wave;                 // grid sized exactly
    int row = (mt << 4) + (lane & 15);
    int hl  = lane >> 4;
    const unsigned short* arow = A + (size_t)row * lda;

    const v8f zero = {0.f, 0.f, 0.f, 0.f, 0.f, 0.f, 0.f, 0.f};
    v8f acc[NTN];
#pragma unroll
    for (int nt = 0; nt < NTN; ++nt) acc[nt] = zero;

    int kTiles = Kp >> 5;
    for (int kb = 0; kb < kTiles; ++kb) {
        Frag a;
        const unsigned short* pa = arow + (kb << 5) + (hl << 4);
        a.q[0] = *(const uint4*)(pa);                // K = 16*hl .. 16*hl+7
        a.q[1] = *(const uint4*)(pa + 8);            // K = 16*hl+8 .. 16*hl+15
#pragma unroll
        for (int nt = 0; nt < NTN; ++nt) {
            Frag b;
            const unsigned short* pb = sW + (((kb * NTN + nt) * 32 + lane) << 4);
            b.q[0] = *(const uint4*)(pb);
            b.q[1] = *(const uint4*)(pb + 8);
            acc[nt] = __builtin_amdgcn_wmma_f32_16x16x32_bf16(
                          false, a.v, false, b.v, (short)0, acc[nt], false, false);
        }
    }

#pragma unroll
    for (int nt = 0; nt < NTN; ++nt) {
        int col = (nt << 4) + (lane & 15);
        float bval = bias ? bias[col] : 0.f;
#pragma unroll
        for (int v = 0; v < 8; ++v) {
            int m = (mt << 4) + v + (hl << 3);
            float r = acc[nt][v] + bval;
            if (relu) r = fmaxf(r, 0.f);
            C[(size_t)m * Nout + col] = r;
        }
    }
}

// ---------------- BatchNorm: batch stats via f32 atomics -------------------
// Linear bias is skipped upstream: BN's mean-subtraction cancels it exactly.
__global__ void k_bnstats(const float* __restrict__ h, float* __restrict__ stats,
                          int M, int C) {
    int c = threadIdx.x;                     // blockDim == C
    float s = 0.f, ss = 0.f;
    for (int r = blockIdx.x; r < M; r += gridDim.x) {
        float v = h[(size_t)r * C + c];
        s += v; ss += v * v;
    }
    atomicAdd(&stats[c], s);
    atomicAdd(&stats[C + c], ss);
}

__global__ void k_bnrelu(const float* __restrict__ h, const float* __restrict__ g,
                         const float* __restrict__ be, const float* __restrict__ stats,
                         unsigned short* __restrict__ out, int M, int C) {
    int t = blockIdx.x * blockDim.x + threadIdx.x;
    if (t >= M * C) return;
    int c = t % C;
    float invM  = 1.0f / (float)M;
    float mean  = stats[c] * invM;
    float var   = stats[C + c] * invM - mean * mean;   // biased var (matches jnp)
    float sc    = g[c] * rsqrtf(var + 1e-5f);
    float sh    = be[c] - mean * sc;
    float v     = fmaf(h[t], sc, sh);
    out[t] = f2bf(fmaxf(v, 0.f));
}

// ---------------- neighbor-sum aggregation ---------------------------------
__global__ void k_aggr(const unsigned short* __restrict__ h, float* __restrict__ outf,
                       unsigned short* __restrict__ outb, int N, int C) {
    int t = blockIdx.x * blockDim.x + threadIdx.x;
    if (t >= N * C) return;
    int n = t / C, c = t - n * C;
    float s = 0.f;
#pragma unroll
    for (int k = 0; k < KNN; ++k)
        s += bf2f(h[((size_t)(n * KNN + k)) * C + c]);
    outf[t] = s;
    if (outb) outb[t] = f2bf(s);
}

// ---------------- head: 256-dot per row (one wave32) + sigmoid -------------
__global__ void k_head(const float* __restrict__ a3, const float* __restrict__ w,
                       const float* __restrict__ b, float* __restrict__ out, int N) {
    int wv   = (int)((blockIdx.x * blockDim.x + threadIdx.x) >> 5);
    int lane = threadIdx.x & 31;
    if (wv >= N) return;
    const float* row = a3 + (size_t)wv * 256;
    float s = 0.f;
#pragma unroll
    for (int e = lane; e < 256; e += 32)
        s = fmaf(row[e], w[e], s);
#pragma unroll
    for (int off = 16; off > 0; off >>= 1)
        s += __shfl_down(s, off, 32);
    if (lane == 0) out[wv] = 1.f / (1.f + __expf(-(s + b[0])));
}

// ===========================================================================
extern "C" void kernel_launch(void* const* d_in, const int* in_sizes, int n_in,
                              void* d_out, int out_size, void* d_ws, size_t ws_size,
                              hipStream_t stream) {
    (void)in_sizes; (void)n_in; (void)out_size; (void)ws_size;
    const int N = NPTS, NK = NEDGE;

    // -------- inputs (setup_inputs dict flatten order) --------
    const float* X     = (const float*)d_in[0];
    // params1: w1 b1 g1 be1 w2 b2 g2 be2 w3 b3 g3 be3   (b* unused: BN cancels bias)
    const float* p1_w1 = (const float*)d_in[1];
    const float* p1_g1 = (const float*)d_in[3];
    const float* p1_be1= (const float*)d_in[4];
    const float* p1_w2 = (const float*)d_in[5];
    const float* p1_g2 = (const float*)d_in[7];
    const float* p1_be2= (const float*)d_in[8];
    const float* p1_w3 = (const float*)d_in[9];
    const float* p1_g3 = (const float*)d_in[11];
    const float* p1_be3= (const float*)d_in[12];
    const float* p2_w1 = (const float*)d_in[13];
    const float* p2_g1 = (const float*)d_in[15];
    const float* p2_be1= (const float*)d_in[16];
    const float* p2_w2 = (const float*)d_in[17];
    const float* p2_g2 = (const float*)d_in[19];
    const float* p2_be2= (const float*)d_in[20];
    const float* p2_w3 = (const float*)d_in[21];
    const float* p2_g3 = (const float*)d_in[23];
    const float* p2_be3= (const float*)d_in[24];
    const float* p3_w1 = (const float*)d_in[25];
    const float* p3_b1 = (const float*)d_in[26];
    const float* p3_w2 = (const float*)d_in[27];
    const float* p3_b2 = (const float*)d_in[28];
    float* out = (float*)d_out;

    // -------- workspace carve (hot set is L2-resident on a 192MB L2) --------
    char* p = (char*)d_ws;
    auto carve = [&](size_t bytes) -> void* {
        void* r = (void*)p; p += (bytes + 255) & ~(size_t)255; return r;
    };
    float*          x2   = (float*)         carve((size_t)N * 4);
    int*            idx  = (int*)           carve((size_t)N * KNN * 4);
    float*          st   = (float*)         carve(2 * 128 * 4);
    unsigned short* Abf  = (unsigned short*)carve((size_t)NK * 128 * 2);  // act bf16
    float*          Hf   = (float*)         carve((size_t)NK * 128 * 4);  // lin out f32
    float*          H1   = (float*)         carve((size_t)N * 64 * 4);
    float*          H2f  = (float*)         carve((size_t)N * 128 * 4);
    unsigned short* H2b  = (unsigned short*)carve((size_t)N * 128 * 2);
    unsigned short* Wt11 = (unsigned short*)carve(32 * 64 * 2);
    unsigned short* Wt12 = (unsigned short*)carve(64 * 64 * 2);
    unsigned short* Wt13 = (unsigned short*)carve(64 * 64 * 2);
    unsigned short* Wt21 = (unsigned short*)carve(128 * 128 * 2);
    unsigned short* Wt22 = (unsigned short*)carve(128 * 128 * 2);
    unsigned short* Wt23 = (unsigned short*)carve(128 * 128 * 2);
    unsigned short* Wt3  = (unsigned short*)carve(128 * 256 * 2);

    auto cdiv = [](int a, int b) { return (a + b - 1) / b; };

    // -------- weight pre-swizzle into WMMA B-fragment tiles --------
    k_prep_w<<<cdiv(32 * 64, 256),   256, 0, stream>>>(p1_w1, Wt11, 16, 32, 64);
    k_prep_w<<<cdiv(64 * 64, 256),   256, 0, stream>>>(p1_w2, Wt12, 64, 64, 64);
    k_prep_w<<<cdiv(64 * 64, 256),   256, 0, stream>>>(p1_w3, Wt13, 64, 64, 64);
    k_prep_w<<<cdiv(128 * 128, 256), 256, 0, stream>>>(p2_w1, Wt21, 128, 128, 128);
    k_prep_w<<<cdiv(128 * 128, 256), 256, 0, stream>>>(p2_w2, Wt22, 128, 128, 128);
    k_prep_w<<<cdiv(128 * 128, 256), 256, 0, stream>>>(p2_w3, Wt23, 128, 128, 128);
    k_prep_w<<<cdiv(128 * 256, 256), 256, 0, stream>>>(p3_w1, Wt3, 128, 128, 256);

    // ======== block 1: edge_conv on X[16384, 8] ========
    k_norms<<<cdiv(N, 256), 256, 0, stream>>>(X, x2, N, 8);
    k_knn<8><<<N / 64, 64, 0, stream>>>(X, x2, idx, N);
    k_edge<<<cdiv(NK * 32, 256), 256, 0, stream>>>(X, idx, Abf, 8, 32);

    struct Layer { const unsigned short* wt; const float* g; const float* be; };
    {
        Layer L[3] = { {Wt11, p1_g1, p1_be1}, {Wt12, p1_g2, p1_be2}, {Wt13, p1_g3, p1_be3} };
        int lda = 32, Kp = 32;
        for (int i = 0; i < 3; ++i) {
            k_gemm<4><<<NK / 128, 256, Kp * 64 * 2, stream>>>(
                Abf, lda, L[i].wt, Hf, nullptr, 0, Kp);
            hipMemsetAsync(st, 0, 2 * 64 * sizeof(float), stream);
            k_bnstats<<<1024, 64, 0, stream>>>(Hf, st, NK, 64);
            k_bnrelu<<<cdiv(NK * 64, 256), 256, 0, stream>>>(
                Hf, L[i].g, L[i].be, st, Abf, NK, 64);
            lda = 64; Kp = 64;
        }
    }
    k_aggr<<<cdiv(N * 64, 256), 256, 0, stream>>>(Abf, H1, nullptr, N, 64);

    // ======== block 2: edge_conv on H1[16384, 64] ========
    k_norms<<<cdiv(N, 256), 256, 0, stream>>>(H1, x2, N, 64);
    k_knn<64><<<N / 64, 64, 0, stream>>>(H1, x2, idx, N);
    k_edge<<<cdiv(NK * 128, 256), 256, 0, stream>>>(H1, idx, Abf, 64, 128);
    {
        Layer L[3] = { {Wt21, p2_g1, p2_be1}, {Wt22, p2_g2, p2_be2}, {Wt23, p2_g3, p2_be3} };
        for (int i = 0; i < 3; ++i) {
            k_gemm<8><<<NK / 128, 256, 128 * 128 * 2, stream>>>(
                Abf, 128, L[i].wt, Hf, nullptr, 0, 128);
            hipMemsetAsync(st, 0, 2 * 128 * sizeof(float), stream);
            k_bnstats<<<1024, 128, 0, stream>>>(Hf, st, NK, 128);
            k_bnrelu<<<cdiv(NK * 128, 256), 256, 0, stream>>>(
                Hf, L[i].g, L[i].be, st, Abf, NK, 128);
        }
    }
    k_aggr<<<cdiv(N * 128, 256), 256, 0, stream>>>(Abf, H2f, H2b, N, 128);

    // ======== head: 128 -> 256 (bias+ReLU, WMMA) -> 1 (sigmoid) ========
    k_gemm<16><<<N / 128, 256, 128 * 256 * 2, stream>>>(
        H2b, 128, Wt3, Hf, p3_b1, 1, 128);
    k_head<<<cdiv(N * 32, 256), 256, 0, stream>>>(Hf, p3_w2, p3_b2, out, N);
}
